// MultiHeadAttention_68221260529857
// MI455X (gfx1250) — matrix operand order repaired and checked by reading, not compiled, not run on hardware
//
#include <hip/hip_runtime.h>
#include <hip/hip_bf16.h>

// Fixed problem dims (from the reference): B=4, S=2048, D=2048, H=16, HD=128.
// Workspace layout (bf16): Qw[B,H,S,HD] | Kw[B,H,S,HD] | Vt[B,H,HD,S] | AO[B,S,D]
//                         | Wqb | Wkb | Wvb | Wob  (each [D,D] bf16)
// => 4*4*16*2048*128*2 + 4*2048*2048*2 bytes = 160 MiB required in d_ws.

constexpr int BDIM   = 4;
constexpr int SEQ    = 2048;
constexpr int DMODEL = 2048;
constexpr int NH     = 16;
constexpr int HDIM   = 128;

constexpr int SPAD = 40;   // LDS row stride in shorts: 80B (16B aligned, bank-friendly)

#if defined(__has_builtin)
#if __has_builtin(__builtin_amdgcn_global_load_async_to_lds_b128)
#define HAVE_ASYNC_LDS 1
#endif
#endif
#ifndef HAVE_ASYNC_LDS
#define HAVE_ASYNC_LDS 0
#endif

typedef __attribute__((ext_vector_type(16))) __bf16 v16bf;
typedef __attribute__((ext_vector_type(8)))  float  v8f;

union FragU { v16bf v; uint4 q[2]; };

__device__ __forceinline__ unsigned short f32_to_bf16(float f) {
  unsigned int u = __float_as_uint(f);
  u += 0x7FFFu + ((u >> 16) & 1u);     // round-to-nearest-even
  return (unsigned short)(u >> 16);
}

// A-matrix 16x32 bf16 fragment: lane = row | (khalf<<4);
// elem e<8 -> K = khalf*8 + e ; e>=8 -> K = 16 + khalf*8 + (e-8)   (ISA 7.12.2)
__device__ __forceinline__ v16bf frag_a(const unsigned short* base, int stride, int lane) {
  const unsigned short* p = base + (size_t)(lane & 15) * stride + (lane >> 4) * 8;
  FragU f;
  f.q[0] = *(const uint4*)(p);
  f.q[1] = *(const uint4*)(p + 16);
  return f.v;
}

// B-matrix 32x16 bf16 fragment: lane = col | (khalf<<4);
// elem e<8 -> K = khalf*8 + e ; e>=8 -> K = 16 + khalf*8 + (e-8)
__device__ __forceinline__ v16bf frag_b(const unsigned short* base, size_t stride, int lane) {
  const unsigned short* p = base + (size_t)(lane & 15) * stride + (lane >> 4) * 8;
  FragU f;
  f.q[0] = *(const uint4*)(p);
  f.q[1] = *(const uint4*)(p + 16);
  return f.v;
}

__device__ __forceinline__ v8f wmma_bf16(v16bf a, v16bf b, v8f c) {
  return __builtin_amdgcn_wmma_f32_16x16x32_bf16(false, a, false, b, (short)0, c,
                                                 false, false);
}

__device__ __forceinline__ float rowmax16(float v) {
#pragma unroll
  for (int m = 1; m <= 8; m <<= 1) v = fmaxf(v, __shfl_xor(v, m, 32));
  return v;
}
__device__ __forceinline__ float rowsum16(float v) {
#pragma unroll
  for (int m = 1; m <= 8; m <<= 1) v += __shfl_xor(v, m, 32);
  return v;
}

__device__ __forceinline__ void cvt8(const float4 a, const float4 b, unsigned short* t) {
  t[0] = f32_to_bf16(a.x); t[1] = f32_to_bf16(a.y);
  t[2] = f32_to_bf16(a.z); t[3] = f32_to_bf16(a.w);
  t[4] = f32_to_bf16(b.x); t[5] = f32_to_bf16(b.y);
  t[6] = f32_to_bf16(b.z); t[7] = f32_to_bf16(b.w);
}

// 16-byte global(bf16) -> LDS copy: async path on CDNA5, register fallback.
#if HAVE_ASYNC_LDS
typedef int async_v4i __attribute__((vector_size(16)));
typedef __attribute__((address_space(1))) async_v4i* g_v4i_p;
typedef __attribute__((address_space(3))) async_v4i* l_v4i_p;
#endif

__device__ __forceinline__ void copy16_g2lds(const unsigned short* gsrc,
                                             unsigned short* ldst) {
#if HAVE_ASYNC_LDS
  void* gs = (void*)gsrc;   // drop const, then reinterpret+addrspacecast
  void* ld = (void*)ldst;
  __builtin_amdgcn_global_load_async_to_lds_b128((g_v4i_p)gs, (l_v4i_p)ld, 0, 0);
#else
  *(uint4*)ldst = *(const uint4*)gsrc;
#endif
}

__device__ __forceinline__ void wait_async0() {
#if HAVE_ASYNC_LDS
#if __has_builtin(__builtin_amdgcn_s_wait_asynccnt)
  __builtin_amdgcn_s_wait_asynccnt(0);
#else
  asm volatile("s_wait_asynccnt 0" ::: "memory");
#endif
#endif
}

// -------------------------------------------------------------------------
// One-time f32 -> bf16 conversion (weights). 8 elems/thread.
// -------------------------------------------------------------------------
__global__ __launch_bounds__(256) void cvt_f32_bf16(const float* __restrict__ in,
                                                    unsigned short* __restrict__ outp,
                                                    int n8) {
  const int i = blockIdx.x * 256 + threadIdx.x;
  if (i < n8) {
    const float4* f4 = (const float4*)(in + (size_t)i * 8);
    __align__(16) unsigned short t[8];
    cvt8(f4[0], f4[1], t);
    *(uint4*)(outp + (size_t)i * 8) = *(uint4*)t;
  }
}

// -------------------------------------------------------------------------
// GEMM: Y = X @ W^T ; X: [M, DMODEL] (TA = float or bf16-as-ushort),
//                     Wb: [DMODEL, DMODEL] bf16 row-major (pre-converted).
// MODE 0: Out f32 [M, DMODEL]
// MODE 1: Out bf16 [B, H, S, HD]   (m -> b,s ; n -> h,hd)
// MODE 2: Out bf16 [B, H, HD, S]   (V transposed for attention B-fragments)
// Block: 128 threads (4 waves), 64x64 tile, each wave 32x32, K-step 32.
// -------------------------------------------------------------------------
template <typename TA, int MODE>
__global__ __launch_bounds__(128) void gemm_xwt(const TA* __restrict__ X,
                                                const unsigned short* __restrict__ Wb,
                                                void* __restrict__ OutP) {
  __shared__ __align__(16) unsigned short sA[64 * SPAD];
  __shared__ __align__(16) unsigned short sB[64 * SPAD];

  const int tid   = threadIdx.x;
  const int lane  = tid & 31;
  const int warp  = tid >> 5;
  const int mBase = blockIdx.y * 64;
  const int nBase = blockIdx.x * 64;
  const int wm    = (warp >> 1) * 32;
  const int wn    = (warp & 1) * 32;

  v8f acc[2][2] = {};

  const int rr = tid >> 2;        // 0..31 : row within half-tile
  const int cc = (tid & 3) * 8;   // 0,8,16,24 : col chunk (8 elems)

  for (int k0 = 0; k0 < DMODEL; k0 += 32) {
    __syncthreads();
    // ---- stage X tile (64 x 32) into LDS as bf16 ----
#pragma unroll
    for (int p = 0; p < 2; ++p) {
      const int r = rr + p * 32;
      if constexpr (sizeof(TA) == 4) {
        __align__(16) unsigned short tmp[8];
        const float4* f4 = (const float4*)(X + (size_t)(mBase + r) * DMODEL + k0 + cc);
        cvt8(f4[0], f4[1], tmp);
        *(uint4*)(sA + r * SPAD + cc) = *(uint4*)tmp;
      } else {
        copy16_g2lds((const unsigned short*)X + (size_t)(mBase + r) * DMODEL + k0 + cc,
                     sA + r * SPAD + cc);
      }
    }
    // ---- stage W tile (bf16 -> async copy to LDS) ----
#pragma unroll
    for (int p = 0; p < 2; ++p) {
      const int r = rr + p * 32;
      copy16_g2lds(Wb + (size_t)(nBase + r) * DMODEL + k0 + cc, sB + r * SPAD + cc);
    }
    // prefetch next K chunk while this tile is consumed
    if (k0 + 32 < DMODEL) {
      __builtin_prefetch(X + (size_t)(mBase + rr) * DMODEL + k0 + 32 + cc, 0, 3);
      __builtin_prefetch(Wb + (size_t)(nBase + rr) * DMODEL + k0 + 32 + cc, 0, 3);
    }
    wait_async0();
    __syncthreads();

    v16bf af[2], bfr[2];
#pragma unroll
    for (int ms = 0; ms < 2; ++ms) af[ms]  = frag_a(sA + (wm + ms * 16) * SPAD, SPAD, lane);
#pragma unroll
    for (int ns = 0; ns < 2; ++ns) bfr[ns] = frag_b(sB + (wn + ns * 16) * SPAD, SPAD, lane);
#pragma unroll
    for (int ms = 0; ms < 2; ++ms)
#pragma unroll
      for (int ns = 0; ns < 2; ++ns)
        acc[ms][ns] = wmma_bf16(af[ms], bfr[ns], acc[ms][ns]);
  }

  // ---- epilogue ----
  const int lc = lane & 15, hf = lane >> 4;
#pragma unroll
  for (int ms = 0; ms < 2; ++ms) {
#pragma unroll
    for (int ns = 0; ns < 2; ++ns) {
      const int n  = nBase + wn + ns * 16 + lc;
      const int m0 = mBase + wm + ms * 16 + hf * 8;
      if constexpr (MODE == 0) {
        float* out = (float*)OutP;
#pragma unroll
        for (int r = 0; r < 8; ++r)
          out[(size_t)(m0 + r) * DMODEL + n] = acc[ms][ns][r];
      } else {
        unsigned short* out = (unsigned short*)OutP;
        const int bb = m0 >> 11;            // / SEQ
        const int s0 = m0 & (SEQ - 1);
        const int hh = n >> 7;              // / HDIM
        const int hd = n & (HDIM - 1);
        if constexpr (MODE == 1) {
          const size_t base = (((size_t)bb * NH + hh) * SEQ + s0) * HDIM + hd;
#pragma unroll
          for (int r = 0; r < 8; ++r)
            out[base + (size_t)r * HDIM] = f32_to_bf16(acc[ms][ns][r]);
        } else {
          const size_t base = (((size_t)bb * NH + hh) * HDIM + hd) * SEQ + s0;
          __align__(16) unsigned short tmp[8];
#pragma unroll
          for (int r = 0; r < 8; ++r) tmp[r] = f32_to_bf16(acc[ms][ns][r]);
          *(uint4*)(out + base) = *(uint4*)tmp;   // s contiguous along rows
        }
      }
    }
  }
}

// -------------------------------------------------------------------------
// Flash attention: each wave owns a 16-row query tile; streams 32 keys/iter.
// Q,K: [B,H,S,HD] bf16 ; Vt: [B,H,HD,S] bf16 ; AO: [B,S,D] bf16.
// Causal + pad mask with -10000 (matches reference). Online softmax.
// -------------------------------------------------------------------------
__global__ __launch_bounds__(128) void flash_attn(
    const unsigned short* __restrict__ Qw, const unsigned short* __restrict__ Kw,
    const unsigned short* __restrict__ Vt, const int* __restrict__ amask,
    unsigned short* __restrict__ AO) {
  __shared__ __align__(16) unsigned short sP[4][16 * SPAD];

  const int lane = threadIdx.x & 31;
  const int warp = threadIdx.x >> 5;
  const int hh   = blockIdx.y;
  const int bb   = blockIdx.z;
  const int q0   = (blockIdx.x * 4 + warp) * 16;   // wave-uniform
  const int lc   = lane & 15;
  const int hf   = lane >> 4;

  const unsigned short* Qh = Qw + (((size_t)bb * NH + hh) * SEQ) * HDIM;
  const unsigned short* Kh = Kw + (((size_t)bb * NH + hh) * SEQ) * HDIM;
  const unsigned short* Vh = Vt + (((size_t)bb * NH + hh) * HDIM) * SEQ;

  // Q A-fragments for the 4 K-chunks of HD=128
  v16bf qf[4];
#pragma unroll
  for (int c = 0; c < 4; ++c) qf[c] = frag_a(Qh + (size_t)q0 * HDIM + c * 32, HDIM, lane);

  v8f o[8] = {};
  float mrow[8], lrow[8];
#pragma unroll
  for (int r = 0; r < 8; ++r) { mrow[r] = -3.0e38f; lrow[r] = 0.0f; }

  const float RSQ = 0.0883883476483184f;   // 1/sqrt(HD)
  const int ktEnd = (q0 + 15) >> 5;        // inclusive; wave-uniform
  unsigned short* sp = sP[warp];

  for (int kt = 0; kt <= ktEnd; ++kt) {
    const int k0 = kt * 32;

    // ---- scores: two 16x16 tiles over 32 keys ----
    v8f s0 = {}, s1 = {};
#pragma unroll
    for (int c = 0; c < 4; ++c) {
      v16bf b0 = frag_b(Kh + (size_t)k0 * HDIM + c * 32, HDIM, lane);
      v16bf b1 = frag_b(Kh + (size_t)(k0 + 16) * HDIM + c * 32, HDIM, lane);
      s0 = wmma_bf16(qf[c], b0, s0);
      s1 = wmma_bf16(qf[c], b1, s1);
    }

    const int kidx0 = k0 + lc;
    const int kidx1 = kidx0 + 16;
    const int mv0 = amask[bb * SEQ + kidx0];
    const int mv1 = amask[bb * SEQ + kidx1];

    // ---- online softmax update ----
    float tscale[8];
#pragma unroll
    for (int r = 0; r < 8; ++r) {
      const int qidx = q0 + r + hf * 8;
      float a0 = s0[r] * RSQ;
      float a1 = s1[r] * RSQ;
      if (kidx0 > qidx || mv0 == 0) a0 = -10000.0f;
      if (kidx1 > qidx || mv1 == 0) a1 = -10000.0f;
      const float tmax = rowmax16(fmaxf(a0, a1));
      const float mnew = fmaxf(mrow[r], tmax);
      const float sc   = __expf(mrow[r] - mnew);
      a0 = __expf(a0 - mnew);
      a1 = __expf(a1 - mnew);
      lrow[r] = lrow[r] * sc + rowsum16(a0 + a1);
      mrow[r] = mnew;
      tscale[r] = sc;
      s0[r] = a0;
      s1[r] = a1;
    }
#pragma unroll
    for (int t = 0; t < 8; ++t)
#pragma unroll
      for (int r = 0; r < 8; ++r) o[t][r] *= tscale[r];

    // ---- transpose P (C-layout -> A-layout) through LDS as bf16 ----
#pragma unroll
    for (int r = 0; r < 8; ++r) {
      const int m = r + hf * 8;
      sp[m * SPAD + lc]      = f32_to_bf16(s0[r]);
      sp[m * SPAD + 16 + lc] = f32_to_bf16(s1[r]);
    }
    asm volatile("s_wait_dscnt 0" ::: "memory");   // RAW across lanes (same wave)
    const v16bf pf = frag_a(sp, SPAD, lane);
    asm volatile("" ::: "memory");

    // ---- O += P @ V  (8 output n-tiles over HD=128) ----
#pragma unroll
    for (int t = 0; t < 8; ++t) {
      v16bf vf = frag_b(Vh + (size_t)(t * 16) * SEQ + k0, SEQ, lane);
      o[t] = wmma_bf16(pf, vf, o[t]);
    }
  }

  // ---- epilogue: O /= l, write [B,S,D] bf16 ----
  float inv[8];
#pragma unroll
  for (int r = 0; r < 8; ++r) inv[r] = 1.0f / lrow[r];
  const size_t obase = ((size_t)bb * SEQ + q0) * DMODEL + (size_t)hh * HDIM;
#pragma unroll
  for (int t = 0; t < 8; ++t) {
    const int hd = t * 16 + lc;
#pragma unroll
    for (int r = 0; r < 8; ++r) {
      const int m = r + hf * 8;
      AO[obase + (size_t)m * DMODEL + hd] = f32_to_bf16(o[t][r] * inv[r]);
    }
  }
}

// -------------------------------------------------------------------------
extern "C" void kernel_launch(void* const* d_in, const int* in_sizes, int n_in,
                              void* d_out, int out_size, void* d_ws, size_t ws_size,
                              hipStream_t stream) {
  (void)in_sizes; (void)n_in; (void)out_size; (void)ws_size;

  const float* q  = (const float*)d_in[0];
  const float* k  = (const float*)d_in[1];
  const float* v  = (const float*)d_in[2];
  const int*   am = (const int*)d_in[3];
  const float* Wq = (const float*)d_in[4];
  const float* Wk = (const float*)d_in[5];
  const float* Wv = (const float*)d_in[6];
  const float* Wo = (const float*)d_in[7];

  const size_t elems = (size_t)BDIM * SEQ * DMODEL;   // 16M elems, bf16 each
  const size_t welems = (size_t)DMODEL * DMODEL;      // 4M elems per weight
  unsigned short* Qw  = (unsigned short*)d_ws;
  unsigned short* Kw  = Qw + elems;
  unsigned short* Vt  = Kw + elems;
  unsigned short* AO  = Vt + elems;
  unsigned short* Wqb = AO + elems;
  unsigned short* Wkb = Wqb + welems;
  unsigned short* Wvb = Wkb + welems;
  unsigned short* Wob = Wvb + welems;

  const dim3 gb(128);
  const dim3 gg(DMODEL / 64, (BDIM * SEQ) / 64);   // (32, 128)
  const int  n8 = (int)(welems / 8);
  const dim3 cg((n8 + 255) / 256), cb(256);

  cvt_f32_bf16<<<cg, cb, 0, stream>>>(Wq, Wqb, n8);
  cvt_f32_bf16<<<cg, cb, 0, stream>>>(Wk, Wkb, n8);
  cvt_f32_bf16<<<cg, cb, 0, stream>>>(Wv, Wvb, n8);
  cvt_f32_bf16<<<cg, cb, 0, stream>>>(Wo, Wob, n8);

  gemm_xwt<float, 1><<<gg, gb, 0, stream>>>(q, Wqb, Qw);
  gemm_xwt<float, 1><<<gg, gb, 0, stream>>>(k, Wkb, Kw);
  gemm_xwt<float, 2><<<gg, gb, 0, stream>>>(v, Wvb, Vt);

  flash_attn<<<dim3(SEQ / 64, NH, BDIM), gb, 0, stream>>>(Qw, Kw, Vt, am, AO);

  gemm_xwt<unsigned short, 0><<<gg, gb, 0, stream>>>(AO, Wob, d_out);
}
